// MultiHeadedAttention_28716151341481
// MI455X (gfx1250) — matrix operand order repaired and checked by reading, not compiled, not run on hardware
//
#include <hip/hip_runtime.h>

typedef __attribute__((ext_vector_type(4)))  _Float16 v4h;
typedef __attribute__((ext_vector_type(8)))  _Float16 v8h;
typedef __attribute__((ext_vector_type(16))) _Float16 v16h;
typedef __attribute__((ext_vector_type(4)))  float    v4f;
typedef __attribute__((ext_vector_type(8)))  float    v8f;
typedef __attribute__((ext_vector_type(4)))  int      v4i;

#define D_MODEL 1024
#define NHEADS  16
#define DK      64
#define BATCH   2
#define SEQ     2048
#define MROWS   (BATCH*SEQ)   // 4096

#define CAT16(lo,hi) __builtin_shufflevector(lo, hi, 0,1,2,3,4,5,6,7,8,9,10,11,12,13,14,15)
#define WMMA(a,b,c) __builtin_amdgcn_wmma_f32_16x16x32_f16(false,(a),false,(b),(short)0,(c),false,false)

// ---- CDNA5 async global->LDS (ASYNCcnt path), guarded with fallback --------
#if __has_builtin(__builtin_amdgcn_global_load_async_to_lds_b128)
#define HAVE_ASYNC_LDS 1
typedef __attribute__((address_space(1))) v4i g_v4i;   // int4 in global AS
typedef __attribute__((address_space(3))) v4i l_v4i;   // int4 in LDS AS
__device__ __forceinline__ void async_copy16(const void* g, void* l) {
  // LDS generic addr: low 32 bits are the LDS offset (aperture rule), so the
  // integer route yields a value-correct as3 pointer without addrspacecast.
  __builtin_amdgcn_global_load_async_to_lds_b128(
      (g_v4i*)(unsigned long long)g,
      (l_v4i*)(unsigned int)(unsigned long long)l, 0, 0);
}
#else
#define HAVE_ASYNC_LDS 0
__device__ __forceinline__ void async_copy16(const void* g, void* l) {
  *(v8h*)l = *(const v8h*)g;
}
#endif

__device__ __forceinline__ void wait_async0() {
#if HAVE_ASYNC_LDS
#if __has_builtin(__builtin_amdgcn_s_wait_asynccnt)
  __builtin_amdgcn_s_wait_asynccnt(0);
#else
  asm volatile("s_wait_asynccnt 0" ::: "memory");
#endif
#endif
}

// ---------------------------------------------------------------------------
// GEMM: C[M,N] = X[M,K] * W[N,K]^T + bias[N]   (torch Linear semantics)
// 128x128 block tile, k-step 32, f16 WMMA with fp32 accumulate.
// IN_HALF:  X is f16 (async-staged);  else fp32 (cvt while staging)
// OUT_HALF: C is f16;                 else fp32
// ---------------------------------------------------------------------------
#define GTM 128
#define GTN 128
#define GTK 32
#define GLD 40   // padded LDS row stride (f16 elems); 80B rows stay 16B aligned

template<bool IN_HALF, bool OUT_HALF>
__global__ __launch_bounds__(256)
void gemm_xwT_bias(const void* __restrict__ Xv, const float* __restrict__ W,
                   const float* __restrict__ bias, void* __restrict__ Cv,
                   int M, int N, int K)
{
  __shared__ __attribute__((aligned(16))) _Float16 Xs[GTM*GLD];
  __shared__ __attribute__((aligned(16))) _Float16 Ws[GTN*GLD];

  const int t      = (int)threadIdx.x;
  const int wid    = t >> 5;
  const int lane   = t & 31;
  const int halfId = lane >> 4;
  const int lr     = lane & 15;

  const int mTiles = M / GTM;
  const int tM = (int)blockIdx.x % mTiles;
  const int tN = (int)blockIdx.x / mTiles;
  const int rowBase = tM * GTM;
  const int colBase = tN * GTN;

  const float*    Xf = (const float*)Xv;
  const _Float16* Xh = (const _Float16*)Xv;

  v8f zero = {};
  v8f acc[8];
#pragma unroll
  for (int n = 0; n < 8; ++n) acc[n] = zero;

  for (int k0 = 0; k0 < K; k0 += GTK) {
    // ---- stage X tile (128x32) ----
    if (IN_HALF) {
#pragma unroll
      for (int i = 0; i < 2; ++i) {
        int c = t + i * 256;          // 0..511 16B chunks
        int r = c >> 2, q = c & 3;    // row, 8-elem group
        async_copy16(Xh + (size_t)(rowBase + r) * K + k0 + q * 8,
                     &Xs[r * GLD + q * 8]);
      }
    } else {
#pragma unroll
      for (int i = 0; i < 4; ++i) {
        int f  = t + i * 256;         // 0..1023
        int r  = f >> 3;              // row 0..127
        int kq = f & 7;               // float4 within k slab
        v4f xv = *(const v4f*)(Xf + (size_t)(rowBase + r) * K + k0 + kq * 4);
        v4h xh = { (_Float16)xv[0], (_Float16)xv[1], (_Float16)xv[2], (_Float16)xv[3] };
        *(v4h*)&Xs[r * GLD + kq * 4] = xh;
      }
    }
    // ---- stage W tile (128x32, fp32 weights -> f16) ----
#pragma unroll
    for (int i = 0; i < 4; ++i) {
      int f  = t + i * 256;
      int r  = f >> 3;
      int kq = f & 7;
      v4f wv = *(const v4f*)(W + (size_t)(colBase + r) * K + k0 + kq * 4);
      v4h wh = { (_Float16)wv[0], (_Float16)wv[1], (_Float16)wv[2], (_Float16)wv[3] };
      *(v4h*)&Ws[r * GLD + kq * 4] = wh;
    }
    // prefetch next k-slab of W (and X when resident fp32) into L2
    if (k0 + GTK < K) {
      __builtin_prefetch(W + (size_t)(colBase + (t >> 1)) * K + k0 + GTK + (t & 1) * 16, 0, 0);
      if (!IN_HALF)
        __builtin_prefetch(Xf + (size_t)(rowBase + (t >> 1)) * K + k0 + GTK + (t & 1) * 16, 0, 0);
    }
    wait_async0();
    __syncthreads();

    // A fragment: this wave's 16 rows, 16x32, ISA A-layout
    {
      int aRow  = wid * 16 + lr;
      int abase = halfId * 8;
      v8h alo = *(const v8h*)&Xs[aRow * GLD + abase];
      v8h ahi = *(const v8h*)&Xs[aRow * GLD + abase + 16];
      v16h afrag = CAT16(alo, ahi);
#pragma unroll
      for (int n = 0; n < 8; ++n) {
        int bRow = n * 16 + lr;           // output column held by this lane
        int bk   = halfId * 8;            // K 8h..8h+7 and 16+8h..16+8h+7 per half-wave
        v8h blo = *(const v8h*)&Ws[bRow * GLD + bk];
        v8h bhi = *(const v8h*)&Ws[bRow * GLD + bk + 16];
        v16h bfrag = CAT16(blo, bhi);
        acc[n] = WMMA(afrag, bfrag, acc[n]);
      }
    }
    __syncthreads();
  }

  // epilogue: C/D layout  M = r + 8*halfId, N = lr
#pragma unroll
  for (int n = 0; n < 8; ++n) {
    int col = colBase + n * 16 + lr;
    float bv = bias[col];
#pragma unroll
    for (int r = 0; r < 8; ++r) {
      int row = rowBase + wid * 16 + r + 8 * halfId;
      float v = acc[n][r] + bv;
      if (OUT_HALF) ((_Float16*)Cv)[(size_t)row * N + col] = (_Float16)v;
      else          ((float*)Cv)[(size_t)row * N + col]    = v;
    }
  }
}

// ---------------------------------------------------------------------------
// Flash-style attention over one (b, h, 128-query block). f16 q/k/v in,
// f16 head-concat out. 8 waves x 16 query rows; stream 64-key blocks.
// ---------------------------------------------------------------------------
#define BR 128
#define BC 64
#define ALD 72   // padded stride for 64 f16 (144B rows, 16B aligned)

__global__ __launch_bounds__(256)
void mha_attention(const _Float16* __restrict__ Q, const _Float16* __restrict__ Km,
                   const _Float16* __restrict__ V, const int* __restrict__ mask,
                   _Float16* __restrict__ O)
{
  __shared__ __attribute__((aligned(16))) _Float16 Ks[BC * ALD];       // [key][dk]
  __shared__ __attribute__((aligned(16))) _Float16 Vt[DK * ALD];       // [dk][key]
  __shared__ __attribute__((aligned(16))) _Float16 Ps[8 * 16 * ALD];   // per-wave P

  const int t      = (int)threadIdx.x;
  const int wid    = t >> 5;
  const int lane   = t & 31;
  const int halfId = lane >> 4;
  const int lr     = lane & 15;

  int blk = (int)blockIdx.x;
  int qt  = blk & 15;            // SEQ/BR = 16 query blocks
  int h   = (blk >> 4) & 15;
  int b   = blk >> 8;
  const int qBase = qt * BR;

  const _Float16* Qh = Q + (size_t)b * SEQ * D_MODEL + h * DK;
  const _Float16* Kh = Km + (size_t)b * SEQ * D_MODEL + h * DK;
  const _Float16* Vh = V + (size_t)b * SEQ * D_MODEL + h * DK;
  const int*      mk = mask + b * SEQ;

  // Q fragments (16 rows x 64 dk as two 16x32 A-fragments), loaded once.
  int qRow = qBase + wid * 16 + lr;
  v16h qf[2];
#pragma unroll
  for (int ks = 0; ks < 2; ++ks) {
    const _Float16* p0 = Qh + (size_t)qRow * D_MODEL + ks * 32 + halfId * 8;
    v8h lo = *(const v8h*)p0;
    v8h hi = *(const v8h*)(p0 + 16);
    qf[ks] = CAT16(lo, hi);
  }

  float m[8], l[8];
  v8f zero = {};
  v8f accO[4];
#pragma unroll
  for (int r = 0; r < 8; ++r) { m[r] = -3.0e38f; l[r] = 0.0f; }
#pragma unroll
  for (int n = 0; n < 4; ++n) accO[n] = zero;

  for (int kb = 0; kb < SEQ / BC; ++kb) {
    int kBase = kb * BC;

    // ---- stage K tile via async global->LDS (pure f16 copy) ----
#pragma unroll
    for (int i = 0; i < 2; ++i) {
      int c = t + i * 256;           // 0..511 16B chunks
      int r = c >> 3, q = c & 7;     // key row, 8-elem group
      async_copy16(Kh + (size_t)(kBase + r) * D_MODEL + q * 8,
                   &Ks[r * ALD + q * 8]);
    }
    // ---- stage V tile transposed (manual: b128 load + b16 scatter) ----
#pragma unroll
    for (int i = 0; i < 2; ++i) {
      int c = t + i * 256;
      int r = c >> 3, q = c & 7;
      v8h vv = *(const v8h*)(Vh + (size_t)(kBase + r) * D_MODEL + q * 8);
#pragma unroll
      for (int j = 0; j < 8; ++j)
        Vt[(q * 8 + j) * ALD + r] = vv[j];
    }
    // prefetch next key block's K and V rows while this block computes
    if (kb + 1 < SEQ / BC) {
      size_t nrow = (size_t)(kBase + BC + (t >> 2)) * D_MODEL + (t & 3) * 16;
      __builtin_prefetch(Kh + nrow, 0, 0);
      __builtin_prefetch(Vh + nrow, 0, 0);
    }
    wait_async0();
    __syncthreads();

    // S = Q * K^T : 4 key col-tiles x 2 k-steps
    v8f accS[4];
#pragma unroll
    for (int n = 0; n < 4; ++n) accS[n] = zero;
#pragma unroll
    for (int n = 0; n < 4; ++n) {
      int kRow = n * 16 + lr;        // key held by this lane (B column)
#pragma unroll
      for (int ks = 0; ks < 2; ++ks) {
        int off = ks * 32 + halfId * 8;
        v8h blo = *(const v8h*)&Ks[kRow * ALD + off];
        v8h bhi = *(const v8h*)&Ks[kRow * ALD + off + 16];
        v16h bf = CAT16(blo, bhi);
        accS[n] = WMMA(qf[ks], bf, accS[n]);
      }
    }

    // scale + mask
#pragma unroll
    for (int n = 0; n < 4; ++n) {
      int kpos = kBase + n * 16 + lr;
      bool ok = (mk[kpos] != 0);
#pragma unroll
      for (int r = 0; r < 8; ++r) {
        float s = accS[n][r] * 0.125f;   // 1/sqrt(64)
        accS[n][r] = ok ? s : -3.0e38f;
      }
    }

    // streaming softmax: half-wave owns one row (C-layout)
#pragma unroll
    for (int r = 0; r < 8; ++r) {
      float mx = fmaxf(fmaxf(accS[0][r], accS[1][r]), fmaxf(accS[2][r], accS[3][r]));
#pragma unroll
      for (int off = 1; off < 16; off <<= 1) mx = fmaxf(mx, __shfl_xor(mx, off, 32));
      float mn = fmaxf(m[r], mx);
      float sc = __expf(m[r] - mn);
      m[r] = mn;
      float su = 0.0f;
#pragma unroll
      for (int n = 0; n < 4; ++n) {
        float p = __expf(accS[n][r] - mn);
        accS[n][r] = p;
        su += p;
      }
#pragma unroll
      for (int off = 1; off < 16; off <<= 1) su += __shfl_xor(su, off, 32);
      l[r] = l[r] * sc + su;
#pragma unroll
      for (int n = 0; n < 4; ++n) accO[n][r] *= sc;
    }

    // P (C-layout) -> per-wave LDS strip (row-major f16), then A-layout reload
    int psBase = wid * 16 * ALD;
#pragma unroll
    for (int n = 0; n < 4; ++n)
#pragma unroll
      for (int r = 0; r < 8; ++r)
        Ps[psBase + (r + 8 * halfId) * ALD + n * 16 + lr] = (_Float16)accS[n][r];
    // same-wave LDS ops stay in order; no barrier needed for Ps

    // O += P * V
#pragma unroll
    for (int ks = 0; ks < 2; ++ks) {
      int abase = ks * 32 + halfId * 8;
      v8h alo = *(const v8h*)&Ps[psBase + lr * ALD + abase];
      v8h ahi = *(const v8h*)&Ps[psBase + lr * ALD + abase + 16];
      v16h af = CAT16(alo, ahi);
#pragma unroll
      for (int n = 0; n < 4; ++n) {
        int vr = (n * 16 + lr) * ALD + ks * 32 + halfId * 8;
        v8h blo = *(const v8h*)&Vt[vr];
        v8h bhi = *(const v8h*)&Vt[vr + 16];
        v16h bf = CAT16(blo, bhi);
        accO[n] = WMMA(af, bf, accO[n]);
      }
    }
    __syncthreads();   // protect Ks/Vt before next block's staging
  }

  // normalize and scatter into [B,S,D] head-concat layout (f16)
#pragma unroll
  for (int n = 0; n < 4; ++n) {
#pragma unroll
    for (int r = 0; r < 8; ++r) {
      int row = qBase + wid * 16 + r + 8 * halfId;
      float val = accO[n][r] / l[r];
      O[((size_t)b * SEQ + row) * D_MODEL + h * DK + n * 16 + lr] = (_Float16)val;
    }
  }
}

// ---------------------------------------------------------------------------
extern "C" void kernel_launch(void* const* d_in, const int* in_sizes, int n_in,
                              void* d_out, int out_size, void* d_ws, size_t ws_size,
                              hipStream_t stream)
{
  const float* query = (const float*)d_in[0];
  const float* key   = (const float*)d_in[1];
  const float* value = (const float*)d_in[2];
  const int*   mask  = (const int*)d_in[3];
  const float* Wq = (const float*)d_in[4];
  const float* bq = (const float*)d_in[5];
  const float* Wk = (const float*)d_in[6];
  const float* bk = (const float*)d_in[7];
  const float* Wv = (const float*)d_in[8];
  const float* bv = (const float*)d_in[9];
  const float* Wo = (const float*)d_in[10];
  const float* bo = (const float*)d_in[11];
  float* out = (float*)d_out;

  _Float16* ws = (_Float16*)d_ws;
  size_t sz = (size_t)MROWS * D_MODEL;
  _Float16* qb = ws;
  _Float16* kb = ws + sz;
  _Float16* vb = ws + 2 * sz;
  _Float16* ab = ws + 3 * sz;

  dim3 blkDim(256);
  dim3 gGrid((MROWS / GTM) * (D_MODEL / GTN));         // 256 blocks
  dim3 aGrid(BATCH * NHEADS * (SEQ / BR));             // 512 blocks

  gemm_xwT_bias<false, true><<<gGrid, blkDim, 0, stream>>>(query, Wq, bq, qb, MROWS, D_MODEL, D_MODEL);
  gemm_xwT_bias<false, true><<<gGrid, blkDim, 0, stream>>>(key,   Wk, bk, kb, MROWS, D_MODEL, D_MODEL);
  gemm_xwT_bias<false, true><<<gGrid, blkDim, 0, stream>>>(value, Wv, bv, vb, MROWS, D_MODEL, D_MODEL);
  mha_attention<<<aGrid, blkDim, 0, stream>>>(qb, kb, vb, mask, ab);
  gemm_xwT_bias<true, false><<<gGrid, blkDim, 0, stream>>>(ab, Wo, bo, out, MROWS, D_MODEL, D_MODEL);
}
